// MaskingModule_59296318488582
// MI455X (gfx1250) — compile-verified, matrix operand order checked by reading, not run on hardware
//
#include <hip/hip_runtime.h>
#include <math.h>

typedef __attribute__((ext_vector_type(16))) _Float16 v16h;
typedef __attribute__((ext_vector_type(8)))  float    v8f;

#define N_       16
#define L_       196
#define D_       1024
#define PPIX     768
#define NBINS    64
#define LEN_KEEP 49
#define NL       (N_ * L_)
#define TOTPIX   (NL * PPIX)

// c2 = (0.5 / sigma^2) * log2(e), sigma = 0.01 ; S = sqrt(c2)
#define C2   7213.4752044f
#define S_   84.93218f
#define EPSR 1e-19f

// d_out layout (floats)
#define X_OFF       0
#define MASK_OFF    (N_ * LEN_KEEP * D_)            // 802816
#define RESTORE_OFF (MASK_OFF + N_ * L_)            // 805952

// workspace layout (4-byte units)
#define WS_MINKEY 0
#define WS_MAXKEY 1
#define WS_ENT    16                 // NL floats
#define WS_IDS    (16 + NL)          // NL ints (ids_shuffle)

// raw hardware transcendentals (no libm denormal fixup)
__device__ __forceinline__ float fast_exp2(float x) { return __builtin_amdgcn_exp2f(x); }
__device__ __forceinline__ float fast_log (float x) { return __builtin_amdgcn_logf(x) * 0.6931471805599453f; }

// ---- order-preserving float <-> uint key (works for any sign) ----
__device__ __forceinline__ unsigned fkey(float f) {
    unsigned u = __float_as_uint(f);
    return (u & 0x80000000u) ? ~u : (u | 0x80000000u);
}
__device__ __forceinline__ float unfkey(unsigned k) {
    unsigned u = (k & 0x80000000u) ? (k & 0x7FFFFFFFu) : ~k;
    return __uint_as_float(u);
}

__device__ __forceinline__ float wave_sum(float v) {
    #pragma unroll
    for (int o = 16; o > 0; o >>= 1) v += __shfl_xor(v, o, 32);
    return v;
}
__device__ __forceinline__ float wave_min(float v) {
    #pragma unroll
    for (int o = 16; o > 0; o >>= 1) v = fminf(v, __shfl_xor(v, o, 32));
    return v;
}
__device__ __forceinline__ float wave_max(float v) {
    #pragma unroll
    for (int o = 16; o > 0; o >>= 1) v = fmaxf(v, __shfl_xor(v, o, 32));
    return v;
}

// ---------------- kernel 0: init min/max keys ----------------
__global__ void k_init(unsigned* ws_u) {
    ws_u[WS_MINKEY] = 0xFFFFFFFFu;
    ws_u[WS_MAXKEY] = 0x00000000u;
}

// ---------------- kernel 1: global min/max of img_pat ----------------
__global__ void k_minmax(const float* __restrict__ img, unsigned* ws_u) {
    __shared__ float smin[8], smax[8];
    float lmin =  3.0e38f, lmax = -3.0e38f;
    const float4* img4 = (const float4*)img;
    const int n4 = TOTPIX / 4;
    for (int i = blockIdx.x * blockDim.x + threadIdx.x; i < n4;
         i += gridDim.x * blockDim.x) {
        float4 v = img4[i];
        lmin = fminf(lmin, fminf(fminf(v.x, v.y), fminf(v.z, v.w)));
        lmax = fmaxf(lmax, fmaxf(fmaxf(v.x, v.y), fmaxf(v.z, v.w)));
    }
    lmin = wave_min(lmin);
    lmax = wave_max(lmax);
    int lane = threadIdx.x & 31, wv = threadIdx.x >> 5;
    if (lane == 0) { smin[wv] = lmin; smax[wv] = lmax; }
    __syncthreads();
    if (threadIdx.x == 0) {
        float bmin = smin[0], bmax = smax[0];
        #pragma unroll
        for (int i = 1; i < 8; ++i) {
            bmin = fminf(bmin, smin[i]);
            bmax = fmaxf(bmax, smax[i]);
        }
        atomicMin(&ws_u[WS_MINKEY], fkey(bmin));
        atomicMax(&ws_u[WS_MAXKEY], fkey(bmax));
    }
}

// ---------------- kernel 2: KDE entropy per patch (WMMA reduction) --------
// One block per (n,l): 256 threads = 8 waves.
// wave w: bin group (w&3)*16..+16, pixel half (w>>2)*384..+384.
// kern value  exp2(-c2*(nv-b)^2) = exp2(-(q - s*b)^2), q = s*nv staged in LDS.
// 16x32 f16 kern tile in V_WMMA_F32_16X16X32_F16 A-layout; B = ones;
// f32 D accumulates the per-bin pixel sums on the matrix unit.
// LDS loads are software-pipelined one WMMA iteration ahead.
__global__ void __launch_bounds__(256)
k_entropy(const float* __restrict__ img, const unsigned* __restrict__ ws_u,
          float* __restrict__ ws_ent) {
    __shared__ float lds_q[PPIX];         // q = s * nv per pixel
    __shared__ float lds_part[2][NBINS];  // per-half raw bin sums (deterministic)

    const int bx   = blockIdx.x;          // n*L + l
    const int tid  = threadIdx.x;
    const int lane = tid & 31;
    const int wv   = tid >> 5;

    const float vmin  = unfkey(ws_u[WS_MINKEY]);
    const float vmax  = unfkey(ws_u[WS_MAXKEY]);
    const float scale = S_ / (vmax - vmin);       // q = (v - vmin) * scale
    const float qoff  = -vmin * scale;

    const float* patch = img + (size_t)bx * PPIX;
    for (int p = tid; p < PPIX; p += 256)
        lds_q[p] = fmaf(patch[p], scale, qoff);
    __syncthreads();

    const int bingrp = wv & 3;            // 16-bin group
    const int half   = wv >> 2;           // pixel half (0/1)
    const int m      = lane & 15;         // A row = bin within group
    const int khi    = lane >> 4;         // A-layout K-offset selector
    const float sb   = (float)(bingrp * 16 + m) * (S_ / 63.0f);  // s * b

    // lane's LDS base: A 16x32 f16 layout gives two contiguous 8-float runs
    // per iteration: [pbase + 8*khi, +8) and [pbase + 16 + 8*khi, +8).
    const float4* qb = (const float4*)&lds_q[half * 384 + (khi << 3)];
    // per iteration 'it' (32 pixels): float4 indices it*8 + {0,1,4,5}

    v16h ones;
    #pragma unroll
    for (int e = 0; e < 16; ++e) ones[e] = (_Float16)1.0f;

    v8f acc = {0.f, 0.f, 0.f, 0.f, 0.f, 0.f, 0.f, 0.f};

    float4 c0 = qb[0], c1 = qb[1], c2 = qb[4], c3 = qb[5];

    #pragma unroll
    for (int it = 0; it < 12; ++it) {
        // prefetch next iteration's q-values into fresh registers
        float4 n0, n1, n2, n3;
        if (it < 11) {
            const int b = (it + 1) * 8;
            n0 = qb[b + 0]; n1 = qb[b + 1]; n2 = qb[b + 4]; n3 = qb[b + 5];
        }
        float qv[16];
        *(float4*)&qv[0]  = c0;
        *(float4*)&qv[4]  = c1;
        *(float4*)&qv[8]  = c2;
        *(float4*)&qv[12] = c3;
        v16h a;
        #pragma unroll
        for (int e = 0; e < 16; ++e) {
            float t = qv[e] - sb;
            a[e] = (_Float16)fast_exp2(-(t * t));   // neg folds into v_exp src mod
        }
        acc = __builtin_amdgcn_wmma_f32_16x16x32_f16(
            false, a, false, ones, (short)0, acc, false, false);
        c0 = n0; c1 = n1; c2 = n2; c3 = n3;
    }

    // D layout: VGPR r, lanes 0-15 -> (M=r, N=lane); lanes 16-31 -> (M=8+r).
    // Column N==0 (lanes 0 and 16) carries each bin's sum; unique writers.
    if ((lane & 15) == 0) {
        const int mb = bingrp * 16 + khi * 8;
        #pragma unroll
        for (int r = 0; r < 8; ++r) lds_part[half][mb + r] = acc[r];
    }
    __syncthreads();

    // entropy on wave 0: 2 bins per lane, deterministic combine of halves
    if (wv == 0) {
        const float inv_p = 1.0f / (float)PPIX;
        float p0 = (lds_part[0][lane]      + lds_part[1][lane])      * inv_p;
        float p1 = (lds_part[0][lane + 32] + lds_part[1][lane + 32]) * inv_p;
        float norm = wave_sum(p0 + p1) + EPSR;
        float q0 = p0 / norm + EPSR;
        float q1 = p1 / norm + EPSR;
        float ent = wave_sum(-(q0 * fast_log(q0) + q1 * fast_log(q1)));
        if (lane == 0) ws_ent[bx] = ent;
    }
}

// ---------------- kernel 3: stable descending rank, mask, restore --------
__global__ void k_rank(const float* __restrict__ ws_ent, int* __restrict__ ws_ids,
                       float* __restrict__ out) {
    __shared__ float sent[L_];
    const int n = blockIdx.x;
    const int i = threadIdx.x;
    if (i < L_) sent[i] = ws_ent[n * L_ + i];
    __syncthreads();
    if (i < L_) {
        const float ei = sent[i];
        int r = 0;
        #pragma unroll 4
        for (int j = 0; j < L_; ++j) {
            float ej = sent[j];
            r += (ej > ei) || (ej == ei && j < i);   // stable argsort(-ent)
        }
        out[RESTORE_OFF + n * L_ + i] = (float)r;    // ids_restore
        out[MASK_OFF    + n * L_ + i] = (r >= LEN_KEEP) ? 1.0f : 0.0f;
        ws_ids[n * L_ + r] = i;                      // ids_shuffle (scatter)
    }
}

// ---------------- kernel 4: gather kept rows (float4) ----------------
__global__ void __launch_bounds__(256)
k_gather(const float* __restrict__ x, const int* __restrict__ ws_ids,
         float* __restrict__ out) {
    const int bx = blockIdx.x;            // n*LEN_KEEP + k
    const int n  = bx / LEN_KEEP;
    const int k  = bx - n * LEN_KEEP;
    const int idx = ws_ids[n * L_ + k];
    const float4* src = (const float4*)(x + ((size_t)(n * L_ + idx)) * D_);
    float4*       dst = (float4*)(out + ((size_t)bx) * D_);
    dst[threadIdx.x] = src[threadIdx.x];  // 256 * float4 = 1024 floats
}

extern "C" void kernel_launch(void* const* d_in, const int* in_sizes, int n_in,
                              void* d_out, int out_size, void* d_ws, size_t ws_size,
                              hipStream_t stream) {
    const float* x   = (const float*)d_in[0];
    const float* img = (const float*)d_in[1];
    float*    out  = (float*)d_out;
    unsigned* ws_u = (unsigned*)d_ws;
    float*    ws_f = (float*)d_ws;

    k_init<<<1, 1, 0, stream>>>(ws_u);
    k_minmax<<<1024, 256, 0, stream>>>(img, ws_u);
    k_entropy<<<NL, 256, 0, stream>>>(img, ws_u, ws_f + WS_ENT);
    k_rank<<<N_, 256, 0, stream>>>(ws_f + WS_ENT, (int*)ws_u + WS_IDS, out);
    k_gather<<<N_ * LEN_KEEP, 256, 0, stream>>>(x, (const int*)ws_u + WS_IDS, out);
}
